// CausalSelfAttention_70978629533868
// MI455X (gfx1250) — compile-verified, matrix operand order checked by reading, not run on hardware
//
#include <hip/hip_runtime.h>

// ---------------------------------------------------------------------------
// CDNA5 (gfx1250) causal self-attention: bf16 WMMA pipeline
//   1) cast x, W_qkv, W_out fp32 -> bf16
//   2) QKV GEMM  (4096x3072x1024) via v_wmma_f32_16x16x32_bf16,
//      B-operand staged through LDS with global_load_async_to_lds_b128
//      (double-buffered, ASYNCcnt-paced)
//   3) V transpose to [B,H,Dh,T]
//   4) flash attention (per-wave 16-row Q tile, 32 keys/step, online softmax)
//   5) output GEMM (4096x1024x1024) -> f32 d_out
// ---------------------------------------------------------------------------

typedef __attribute__((ext_vector_type(16))) __bf16 v16bf;
typedef __attribute__((ext_vector_type(8)))  __bf16 v8bf;
typedef __attribute__((ext_vector_type(8)))  float  v8f;
typedef __attribute__((ext_vector_type(4)))  int    v4i;

#define B_SZ   2
#define T_SZ   2048
#define C_SZ   1024
#define H_SZ   16
#define DH_SZ  64

#if defined(__gfx1250__) && \
    __has_builtin(__builtin_amdgcn_global_load_async_to_lds_b128) && \
    __has_builtin(__builtin_amdgcn_s_wait_asynccnt)
#define USE_ASYNC 1
#define ASYNC_WAIT(n) __builtin_amdgcn_s_wait_asynccnt(n)
typedef __attribute__((address_space(1))) v4i* gbl_v4i_ptr;
typedef __attribute__((address_space(3))) v4i* lds_v4i_ptr;
#else
#define USE_ASYNC 0
#define ASYNC_WAIT(n)
#endif

__device__ __forceinline__ v8f wmma_bf16(v16bf a, v16bf b, v8f c) {
  // v_wmma_f32_16x16x32_bf16: (neg_a, A, neg_b, B, c_mod, C, reuse_a, reuse_b)
  return __builtin_amdgcn_wmma_f32_16x16x32_bf16(false, a, false, b,
                                                 (short)0, c, false, false);
}

// Load one 16-lane-half fragment: 8 contiguous bf16 at p, 8 more at p+16.
// Matches ISA 16-bit A/B layout: lane-half h covers K = {h*8..h*8+7, h*8+16..h*8+23}
// within a 32-wide K window (caller passes p = row + kwindow + h*8).
__device__ __forceinline__ v16bf load_frag(const __bf16* p) {
  v8bf lo = *(const v8bf*)(p);
  v8bf hi = *(const v8bf*)(p + 16);
  v16bf r;
#pragma unroll
  for (int i = 0; i < 8; ++i) { r[i] = lo[i]; r[i + 8] = hi[i]; }
  return r;
}

// ---------------------------------------------------------------------------
__global__ void f32_to_bf16_kernel(const float* __restrict__ src,
                                   __bf16* __restrict__ dst, int n) {
  int i = blockIdx.x * 256 + threadIdx.x;
  if (i < n) dst[i] = (__bf16)src[i];
}

// ---------------------------------------------------------------------------
// C[m,n] = sum_k A[m,k] * Bw[n,k]   (A: [M,K] row-major, Bw: weight [N,K])
// 128 threads = 4 waves; each wave -> 16(M) x 64(N); block -> 64x64.
// B 64x32 k-slice is staged once per block into LDS (async, double-buffered);
// rows padded to 40 bf16 (80 B = 20 banks) -> conflict-free b128 reads.
template <bool BF16OUT>
__global__ __launch_bounds__(128) void gemm_wmma_kernel(
    const __bf16* __restrict__ A, const __bf16* __restrict__ Bw,
    void* __restrict__ Cout, int M, int N, int K, int ldc) {
  constexpr int BP = 40;  // padded LDS row length (bf16 elements)
  __shared__ __align__(16) __bf16 btile[2][64][BP];

  const int tid  = threadIdx.x;
  const int lane = tid & 31;
  const int wv   = tid >> 5;
  const int r    = lane & 15;
  const int hf   = lane >> 4;

  const int mbase = blockIdx.y * 64 + wv * 16;
  const int nbase = blockIdx.x * 64;

  const __bf16* arow = A + (size_t)(mbase + r) * K + hf * 8;

  // Stage the 64(N) x 32(K) weight slice at k-window `kb` into btile[buf].
  // 256 x 16B chunks, 2 per thread; each wave issues 2 async instructions.
  auto stage = [&](int buf, int kb) {
#pragma unroll
    for (int c = 0; c < 2; ++c) {
      int chunk = tid * 2 + c;          // 0..255
      int row   = chunk >> 2;           // 0..63
      int kc    = (chunk & 3) * 8;      // 0,8,16,24 (bf16 elems; 16B steps)
      const __bf16* g = Bw + (size_t)(nbase + row) * K + kb + kc;
#if USE_ASYNC
      __builtin_amdgcn_global_load_async_to_lds_b128(
          (gbl_v4i_ptr)(g),
          (lds_v4i_ptr)(&btile[buf][row][kc]),
          0, 0);
#else
      *(v8bf*)&btile[buf][row][kc] = *(const v8bf*)g;
#endif
    }
  };

  v8f acc[4] = {};

  auto compute = [&](int buf, int kb) {
    __builtin_prefetch(arow + kb + 128, 0, 0);
    v16bf a = load_frag(arow + kb);     // A direct from global (L2-resident)
#pragma unroll
    for (int nt = 0; nt < 4; ++nt) {
      v8bf lo = *(const v8bf*)&btile[buf][nt * 16 + r][hf * 8];
      v8bf hi = *(const v8bf*)&btile[buf][nt * 16 + r][hf * 8 + 16];
      v16bf b;
#pragma unroll
      for (int i = 0; i < 8; ++i) { b[i] = lo[i]; b[i + 8] = hi[i]; }
      acc[nt] = wmma_bf16(a, b, acc[nt]);
    }
  };

  stage(0, 0);
  int cur = 0, kb = 0;
  for (; kb + 32 < K; kb += 32) {
    stage(cur ^ 1, kb + 32);   // prefetch next slice into the other buffer
    ASYNC_WAIT(2);             // oldest 2 async ops (current buffer) complete
    __syncthreads();
    compute(cur, kb);
    __syncthreads();           // all reads of btile[cur] done before reuse
    cur ^= 1;
  }
  ASYNC_WAIT(0);
  __syncthreads();
  compute(cur, kb);

  // C/D layout: vgpr v, lanes 0-15: (m=v, n=lane); lanes 16-31: (m=v+8, n=lane-16)
#pragma unroll
  for (int nt = 0; nt < 4; ++nt) {
#pragma unroll
    for (int v = 0; v < 8; ++v) {
      int row = mbase + v + 8 * hf;
      int col = nbase + nt * 16 + r;
      if constexpr (BF16OUT)
        ((__bf16*)Cout)[(size_t)row * ldc + col] = (__bf16)acc[nt][v];
      else
        ((float*)Cout)[(size_t)row * ldc + col] = acc[nt][v];
    }
  }
}

// ---------------------------------------------------------------------------
// vT[b][h][d][t] = qkv[b][t][2C + h*64 + d]
__global__ void transpose_v_kernel(const __bf16* __restrict__ qkv,
                                   __bf16* __restrict__ vT) {
  int idx = blockIdx.x * 256 + threadIdx.x;           // 0 .. B*T*C-1
  int c  = idx & (C_SZ - 1);
  int bt = idx >> 10;
  int t  = bt & (T_SZ - 1);
  int b  = bt >> 11;
  int h  = c >> 6;
  int d  = c & 63;
  vT[(((size_t)b * H_SZ + h) * DH_SZ + d) * T_SZ + t] =
      qkv[(size_t)bt * (3 * C_SZ) + 2 * C_SZ + c];
}

// ---------------------------------------------------------------------------
// Flash attention. grid = (T/64, B*H), block = 128 (4 waves, 16 q-rows each).
__global__ __launch_bounds__(128) void attn_kernel(
    const __bf16* __restrict__ qkv, const __bf16* __restrict__ vT,
    __bf16* __restrict__ attn_out) {
  const int lane = threadIdx.x & 31;
  const int wv   = threadIdx.x >> 5;
  const int r    = lane & 15;
  const int hf   = lane >> 4;

  const int qblk = blockIdx.x;
  const int bh   = blockIdx.y;
  const int b    = bh >> 4;
  const int h    = bh & 15;
  const int qbase = qblk * 64 + wv * 16;

  __shared__ __align__(16) __bf16 plds[4][16][32];  // per-wave P staging

  const __bf16* qptr = qkv + (size_t)b * T_SZ * (3 * C_SZ) + h * DH_SZ;
  const __bf16* kptr = qptr + C_SZ;
  const __bf16* vtb  = vT + ((size_t)(b * H_SZ + h)) * DH_SZ * T_SZ;

  // Q tile (16x64) -> two K=32 A-fragments, kept in registers
  v16bf qf[2];
  {
    const __bf16* rowp = qptr + (size_t)(qbase + r) * (3 * C_SZ);
#pragma unroll
    for (int f = 0; f < 2; ++f) qf[f] = load_frag(rowp + f * 32 + hf * 8);
  }

  v8f acc[4] = {};
  float mrow[8], lrow[8];
#pragma unroll
  for (int v = 0; v < 8; ++v) { mrow[v] = -1.0e30f; lrow[v] = 0.0f; }

  const float scale = 0.125f;  // 1/sqrt(64)
  const int qlast = qbase + 15;

  for (int s0 = 0; s0 <= qlast; s0 += 32) {
    // ---- S = Q K^T for 32 keys (two 16-key C tiles) ----
    v8f st[2];
#pragma unroll
    for (int j = 0; j < 2; ++j) {
      const __bf16* krow = kptr + (size_t)(s0 + j * 16 + r) * (3 * C_SZ);
      v8f c = {};
      c = wmma_bf16(qf[0], load_frag(krow + 0  + hf * 8), c);
      c = wmma_bf16(qf[1], load_frag(krow + 32 + hf * 8), c);
      st[j] = c;
    }

    // ---- scale + causal mask + row max ----
    float pv0[8], pv1[8], mx[8];
#pragma unroll
    for (int v = 0; v < 8; ++v) {
      int qrow = qbase + v + 8 * hf;
      float a0 = (float)st[0][v] * scale;
      float a1 = (float)st[1][v] * scale;
      if (s0 + r      > qrow) a0 = -1.0e30f;
      if (s0 + 16 + r > qrow) a1 = -1.0e30f;
      pv0[v] = a0; pv1[v] = a1;
      mx[v] = fmaxf(a0, a1);
    }
#pragma unroll
    for (int v = 0; v < 8; ++v)
#pragma unroll
      for (int off = 1; off < 16; off <<= 1)
        mx[v] = fmaxf(mx[v], __shfl_xor(mx[v], off, 16));

    // ---- online softmax update ----
    float alpha[8], psum[8];
#pragma unroll
    for (int v = 0; v < 8; ++v) {
      float mn = fmaxf(mrow[v], mx[v]);
      alpha[v] = __expf(mrow[v] - mn);
      mrow[v]  = mn;
      float p0 = __expf(pv0[v] - mn);
      float p1 = __expf(pv1[v] - mn);
      pv0[v] = p0; pv1[v] = p1;
      psum[v] = p0 + p1;
    }
#pragma unroll
    for (int v = 0; v < 8; ++v)
#pragma unroll
      for (int off = 1; off < 16; off <<= 1)
        psum[v] += __shfl_xor(psum[v], off, 16);
#pragma unroll
    for (int v = 0; v < 8; ++v) lrow[v] = lrow[v] * alpha[v] + psum[v];
#pragma unroll
    for (int nt = 0; nt < 4; ++nt)
#pragma unroll
      for (int v = 0; v < 8; ++v) acc[nt][v] *= alpha[v];

    // ---- P: C-layout -> A-layout via per-wave LDS (same-wave, in order) ----
#pragma unroll
    for (int v = 0; v < 8; ++v) {
      plds[wv][v + 8 * hf][r]      = (__bf16)pv0[v];
      plds[wv][v + 8 * hf][16 + r] = (__bf16)pv1[v];
    }
    v16bf pf;
    {
      v8bf lo = *(const v8bf*)&plds[wv][r][hf * 8];
      v8bf hi = *(const v8bf*)&plds[wv][r][hf * 8 + 16];
#pragma unroll
      for (int i = 0; i < 8; ++i) { pf[i] = lo[i]; pf[i + 8] = hi[i]; }
    }

    // ---- acc += P (16x32) @ V (32x64), V^T rows contiguous ----
#pragma unroll
    for (int dt = 0; dt < 4; ++dt) {
      v16bf vf = load_frag(vtb + (size_t)(dt * 16 + r) * T_SZ + s0 + hf * 8);
      acc[dt] = wmma_bf16(pf, vf, acc[dt]);
    }
  }

  // ---- epilogue: normalize rows, store bf16 [B,T,C] at column h*64 ----
#pragma unroll
  for (int dt = 0; dt < 4; ++dt) {
#pragma unroll
    for (int v = 0; v < 8; ++v) {
      int t   = qbase + v + 8 * hf;
      int col = h * DH_SZ + dt * 16 + r;
      attn_out[((size_t)b * T_SZ + t) * C_SZ + col] =
          (__bf16)(acc[dt][v] / lrow[v]);
    }
  }
}

// ---------------------------------------------------------------------------
extern "C" void kernel_launch(void* const* d_in, const int* in_sizes, int n_in,
                              void* d_out, int out_size, void* d_ws, size_t ws_size,
                              hipStream_t stream) {
  const float* x    = (const float*)d_in[0];   // [2,2048,1024]
  const float* Wqkv = (const float*)d_in[1];   // [3072,1024]
  const float* Wout = (const float*)d_in[2];   // [1024,1024]
  float* out = (float*)d_out;                  // [2,2048,1024]

  char* ws = (char*)d_ws;
  const size_t MB = 1u << 20;
  __bf16* xb    = (__bf16*)(ws +  0 * MB);  // 8 MB : x bf16
  __bf16* wqkvb = (__bf16*)(ws +  8 * MB);  // 6 MB : W_qkv bf16
  __bf16* woutb = (__bf16*)(ws + 14 * MB);  // 2 MB : W_out bf16
  __bf16* qkv   = (__bf16*)(ws + 16 * MB);  // 24 MB: qkv [B,T,3C] bf16
  __bf16* vT    = (__bf16*)(ws + 40 * MB);  // 8 MB : V^T [B,H,Dh,T] bf16
  __bf16* attn  = (__bf16*)(ws + 48 * MB);  // 8 MB : attn out [B,T,C] bf16

  const int n_x    = B_SZ * T_SZ * C_SZ;       // 4,194,304
  const int n_wqkv = 3 * C_SZ * C_SZ;          // 3,145,728
  const int n_wout = C_SZ * C_SZ;              // 1,048,576

  f32_to_bf16_kernel<<<(n_x + 255) / 256, 256, 0, stream>>>(x, xb, n_x);
  f32_to_bf16_kernel<<<(n_wqkv + 255) / 256, 256, 0, stream>>>(Wqkv, wqkvb, n_wqkv);
  f32_to_bf16_kernel<<<(n_wout + 255) / 256, 256, 0, stream>>>(Wout, woutb, n_wout);

  // QKV projection: [4096,1024] x [3072,1024]^T -> [4096,3072] bf16
  gemm_wmma_kernel<true><<<dim3(3072 / 64, 4096 / 64), 128, 0, stream>>>(
      xb, wqkvb, (void*)qkv, 4096, 3072, 1024, 3072);

  // V transpose -> [B,H,Dh,T]
  transpose_v_kernel<<<n_x / 256, 256, 0, stream>>>(qkv, vT);

  // Flash attention
  attn_kernel<<<dim3(T_SZ / 64, B_SZ * H_SZ), 128, 0, stream>>>(qkv, vT, attn);

  // Output projection: [4096,1024] x [1024,1024]^T -> [4096,1024] f32
  gemm_wmma_kernel<false><<<dim3(1024 / 64, 4096 / 64), 128, 0, stream>>>(
      attn, woutb, (void*)out, 4096, 1024, 1024, 1024);
}
